// ClusterModel_6451040879217
// MI455X (gfx1250) — compile-verified, matrix operand order checked by reading, not run on hardware
//
#include <hip/hip_runtime.h>
#include <hip/hip_bf16.h>
#include <cstdint>
#include <cstddef>

// ---------------------------------------------------------------------------
// CDNA5 (gfx1250) wave32 WMMA point-transformer forward pass.
// LDS layouts arranged so every WMMA fragment is two ds_load_b128 per lane;
// global->LDS staging is vectorized/batched so loads issue without per-element
// waits.
// ---------------------------------------------------------------------------

typedef __attribute__((ext_vector_type(16))) _Float16 v16h;
typedef __attribute__((ext_vector_type(8)))  _Float16 v8h;
typedef __attribute__((ext_vector_type(4)))  _Float16 v4h;
typedef __attribute__((ext_vector_type(8)))  float    v8f;

#define NBATCH 2
#define HEADS  8
#define KNN    16

// Per-lane WMMA 16-bit fragment: elements 0..7 -> K = k0..k0+7,
// elements 8..15 -> K = k0+16..k0+23  (k0 = half*8).  Both runs contiguous
// in a row-major [row][K] LDS tile -> two 16-byte LDS loads.
__device__ __forceinline__ v16h frag16(const _Float16* p, int k0)
{
  union { v16h v; v8h h[2]; } u;
  u.h[0] = *(const v8h*)(p + k0);
  u.h[1] = *(const v8h*)(p + k0 + 16);
  return u.v;
}

__device__ __forceinline__ v8h cvt8(float4 a, float4 b)
{
  v8h h;
  h[0] = (_Float16)a.x; h[1] = (_Float16)a.y; h[2] = (_Float16)a.z; h[3] = (_Float16)a.w;
  h[4] = (_Float16)b.x; h[5] = (_Float16)b.y; h[6] = (_Float16)b.z; h[7] = (_Float16)b.w;
  return h;
}

// ===========================================================================
// Generic GEMM:  C[M,N] = act( A[M,K] @ W[K,N] + bias[N] )  (optional +=)
// A,W,C row-major f32; staged to LDS as f16; v_wmma_f32_16x16x32_f16.
// M % 64 == 0, N % 64 == 0 guaranteed by caller; K arbitrary (zero-pad).
// Block: 256 threads = 8 waves; block tile 64x64; wave tile 16x32.
// B stored TRANSPOSED in LDS (Bt[n][k]) so B-fragments are contiguous.
// Fast staging path (no guards, float4 loads) when K % 32 == 0 — true for
// every large GEMM in this net; only K=3 (pos-enc) and K=16 (tail) fall back.
// ===========================================================================
__global__ __launch_bounds__(256) void k_gemm(
    const float* __restrict__ A, const float* __restrict__ W,
    const float* __restrict__ bias, float* __restrict__ C,
    int M, int N, int K, int act, int accum)
{
  __shared__ alignas(16) _Float16 As[64][40];   // [m][k]
  __shared__ alignas(16) _Float16 Bt[64][40];   // [n][k]  (transposed)
  const int tid  = threadIdx.x;
  const int wave = tid >> 5;
  const int lane = tid & 31;
  const int lr   = lane & 15;
  const int half = lane >> 4;
  const int m0 = blockIdx.x * 64;
  const int n0 = blockIdx.y * 64;
  const int mw = (wave & 3) * 16;   // wave row sub-block
  const int nw = (wave >> 2) * 32;  // wave col sub-block (two 16-wide tiles)
  const int k0 = half * 8;
  const bool kfast = ((K & 31) == 0);   // uniform branch

  v8f c0 = {}; v8f c1 = {};

  for (int kk = 0; kk < K; kk += 32) {
    if (kfast) {
      // ---- A: two float4 loads -> one 16B ds_store_b128 ----
      {
        int r = tid >> 2, kq = (tid & 3) * 8;
        const float* ap = &A[(size_t)(m0 + r) * K + kk + kq];
        float4 v0 = *(const float4*)(ap);
        float4 v1 = *(const float4*)(ap + 4);
        *(v8h*)&As[r][kq] = cvt8(v0, v1);
      }
      // ---- B: two float4 loads -> 8 transposed b16 stores ----
      {
        int r = tid >> 3, c8 = (tid & 7) * 8;
        const float* wp = &W[(size_t)(kk + r) * N + n0 + c8];
        float4 v0 = *(const float4*)(wp);
        float4 v1 = *(const float4*)(wp + 4);
        Bt[c8 + 0][r] = (_Float16)v0.x; Bt[c8 + 1][r] = (_Float16)v0.y;
        Bt[c8 + 2][r] = (_Float16)v0.z; Bt[c8 + 3][r] = (_Float16)v0.w;
        Bt[c8 + 4][r] = (_Float16)v1.x; Bt[c8 + 5][r] = (_Float16)v1.y;
        Bt[c8 + 6][r] = (_Float16)v1.z; Bt[c8 + 7][r] = (_Float16)v1.w;
      }
    } else {
      // ---- slow path: batch loads first (8 outstanding), then store ----
      float av[8];
#pragma unroll
      for (int j = 0; j < 8; ++j) {
        int i = tid + j * 256;
        int r = i >> 5, k = i & 31;
        int kg = kk + k;
        av[j] = A[(size_t)(m0 + r) * K + ((kg < K) ? kg : (K - 1))];
      }
#pragma unroll
      for (int j = 0; j < 8; ++j) {
        int i = tid + j * 256;
        int r = i >> 5, k = i & 31;
        As[r][k] = (_Float16)(((kk + k) < K) ? av[j] : 0.f);
      }
      float bv[8];
#pragma unroll
      for (int j = 0; j < 8; ++j) {
        int i = tid + j * 256;
        int r = i >> 6, c = i & 63;
        int kg = kk + r;
        bv[j] = W[(size_t)((kg < K) ? kg : (K - 1)) * N + n0 + c];
      }
#pragma unroll
      for (int j = 0; j < 8; ++j) {
        int i = tid + j * 256;
        int r = i >> 6, c = i & 63;
        Bt[c][r] = (_Float16)(((kk + r) < K) ? bv[j] : 0.f);
      }
    }
    // prefetch next K-tile (global_prefetch_b8)
    if (kk + 32 < K) {
      __builtin_prefetch(&A[(size_t)(m0 + (tid >> 2)) * K + kk + 32 + (tid & 3) * 8], 0, 0);
      __builtin_prefetch(&W[(size_t)(kk + 32 + (tid >> 3)) * N + n0 + (tid & 7) * 8], 0, 0);
    }
    __syncthreads();

    v16h a  = frag16(&As[mw + lr][0],      k0);
    v16h b0 = frag16(&Bt[nw + lr][0],      k0);
    v16h b1 = frag16(&Bt[nw + 16 + lr][0], k0);
    c0 = __builtin_amdgcn_wmma_f32_16x16x32_f16(false, a, false, b0, (short)0, c0, false, false);
    c1 = __builtin_amdgcn_wmma_f32_16x16x32_f16(false, a, false, b1, (short)0, c1, false, false);
    __syncthreads();
  }

  const int gn0 = n0 + nw + lr;
  const int gn1 = gn0 + 16;
  const float bi0 = bias ? bias[gn0] : 0.f;
  const float bi1 = bias ? bias[gn1] : 0.f;
#pragma unroll
  for (int j = 0; j < 8; ++j) {
    int gm = m0 + mw + j + half * 8;
    float v0 = c0[j] + bi0, v1 = c1[j] + bi1;
    if (act == 1)      { v0 = fmaxf(v0, 0.f); v1 = fmaxf(v1, 0.f); }
    else if (act == 2) { v0 = v0 / (1.f + __expf(-v0)); v1 = v1 / (1.f + __expf(-v1)); }
    size_t i0 = (size_t)gm * N + gn0, i1 = (size_t)gm * N + gn1;
    if (accum) { C[i0] += v0; C[i1] += v1; }
    else       { C[i0]  = v0; C[i1]  = v1; }
  }
}

// ===========================================================================
// Fused top-level FFN:  Y = relu(X @ W1 + b1) @ W2 + b2,  E=64, F=2048.
// Avoids the 512 MB hidden-activation round trip: the hidden tile lives in
// LDS in 32-column chunks, produced and consumed by WMMA inside one block.
// W1/W2 stored transposed in LDS so all fragments are ds_load_b128 pairs.
// ===========================================================================
__global__ __launch_bounds__(256) void k_ffn_top(
    const float* __restrict__ X,
    const float* __restrict__ W1, const float* __restrict__ b1,
    const float* __restrict__ W2, const float* __restrict__ b2,
    float* __restrict__ Y, int M)
{
  const int E = 64, F = 2048;
  __shared__ alignas(16) _Float16 Xs[64][72];    // [m][k]   (K=64)
  __shared__ alignas(16) _Float16 W1t[32][72];   // [f][k]   (transposed)
  __shared__ alignas(16) _Float16 Hs[64][40];    // [m][f32chunk]
  __shared__ alignas(16) _Float16 W2t[64][40];   // [n][k]   (transposed)

  const int tid = threadIdx.x, wave = tid >> 5, lane = tid & 31;
  const int lr = lane & 15, half = lane >> 4;
  const int m0 = blockIdx.x * 64;
  const int mw = (wave & 3) * 16;
  const int nw = (wave >> 2) * 32;
  const int k0 = half * 8;

  // stage X tile: batch 4 float4 loads, then 4 v4h stores
  {
    float4 xv[4];
#pragma unroll
    for (int j = 0; j < 4; ++j) {
      int i = tid + j * 256;
      int r = i >> 4, c4 = (i & 15) * 4;
      xv[j] = *(const float4*)&X[(size_t)(m0 + r) * E + c4];
    }
#pragma unroll
    for (int j = 0; j < 4; ++j) {
      int i = tid + j * 256;
      int r = i >> 4, c4 = (i & 15) * 4;
      v4h h;
      h[0] = (_Float16)xv[j].x; h[1] = (_Float16)xv[j].y;
      h[2] = (_Float16)xv[j].z; h[3] = (_Float16)xv[j].w;
      *(v4h*)&Xs[r][c4] = h;
    }
  }

  v8f y0 = {}, y1 = {};

  for (int f0 = 0; f0 < F; f0 += 32) {
    // stage W1 chunk transposed: W1t[f][k] = W1[k, f0+f]  (2 float4 / thread)
    {
      float4 wv[2];
#pragma unroll
      for (int j = 0; j < 2; ++j) {
        int i = tid + j * 256;
        int r = i >> 3, c4 = (i & 7) * 4;                // r = k, c4 = f
        wv[j] = *(const float4*)&W1[(size_t)r * F + f0 + c4];
      }
#pragma unroll
      for (int j = 0; j < 2; ++j) {
        int i = tid + j * 256;
        int r = i >> 3, c4 = (i & 7) * 4;
        W1t[c4 + 0][r] = (_Float16)wv[j].x;
        W1t[c4 + 1][r] = (_Float16)wv[j].y;
        W1t[c4 + 2][r] = (_Float16)wv[j].z;
        W1t[c4 + 3][r] = (_Float16)wv[j].w;
      }
    }
    __syncthreads();  // A: prior-iter WMMA reads done; Xs/W1t visible

    // H tile (64x32) = relu(Xs @ W1 + b1): 8 waves, one 16x16 tile each.
    {
      const int hm = (wave & 3) * 16, hn = (wave >> 2) * 16;
      v8f hc = {};
#pragma unroll
      for (int kk = 0; kk < 64; kk += 32) {
        v16h a = frag16(&Xs[hm + lr][0],  kk + k0);
        v16h b = frag16(&W1t[hn + lr][0], kk + k0);
        hc = __builtin_amdgcn_wmma_f32_16x16x32_f16(false, a, false, b, (short)0, hc, false, false);
      }
      __syncthreads();  // B
      float bb = b1[f0 + hn + lr];
#pragma unroll
      for (int j = 0; j < 8; ++j)
        Hs[hm + j + half * 8][hn + lr] = (_Float16)fmaxf(hc[j] + bb, 0.f);
    }
    // stage W2 chunk transposed: W2t[n][k] = W2[f0+k, n]  (2 float4 / thread)
    {
      float4 wv[2];
#pragma unroll
      for (int j = 0; j < 2; ++j) {
        int i = tid + j * 256;
        int r = i >> 4, c4 = (i & 15) * 4;               // r = k, c4 = n
        wv[j] = *(const float4*)&W2[(size_t)(f0 + r) * E + c4];
      }
#pragma unroll
      for (int j = 0; j < 2; ++j) {
        int i = tid + j * 256;
        int r = i >> 4, c4 = (i & 15) * 4;
        W2t[c4 + 0][r] = (_Float16)wv[j].x;
        W2t[c4 + 1][r] = (_Float16)wv[j].y;
        W2t[c4 + 2][r] = (_Float16)wv[j].z;
        W2t[c4 + 3][r] = (_Float16)wv[j].w;
      }
    }
    if (f0 + 32 < F) {
      __builtin_prefetch(&W1[(size_t)(tid >> 2) * F + f0 + 32 + (tid & 3) * 8], 0, 0);
      __builtin_prefetch(&W2[(size_t)(f0 + 32 + (tid >> 3)) * E + (tid & 7) * 8], 0, 0);
    }
    __syncthreads();  // C: Hs/W2t visible

    {
      v16h a   = frag16(&Hs[mw + lr][0],       k0);
      v16h b0  = frag16(&W2t[nw + lr][0],      k0);
      v16h b1v = frag16(&W2t[nw + 16 + lr][0], k0);
      y0 = __builtin_amdgcn_wmma_f32_16x16x32_f16(false, a, false, b0,  (short)0, y0, false, false);
      y1 = __builtin_amdgcn_wmma_f32_16x16x32_f16(false, a, false, b1v, (short)0, y1, false, false);
    }
    __syncthreads();
  }

  const int gn0 = nw + lr, gn1 = gn0 + 16;
  const float bb0 = b2[gn0], bb1 = b2[gn1];
#pragma unroll
  for (int j = 0; j < 8; ++j) {
    int gm = m0 + mw + j + half * 8;
    Y[(size_t)gm * E + gn0] = y0[j] + bb0;
    Y[(size_t)gm * E + gn1] = y1[j] + bb1;
  }
}

// ===========================================================================
// KNN: unordered top-16 smallest squared distances. All consumers are
// order-invariant across K (softmax+mean, max-pool, scatter-mean).
// idx layout: (K, Lq, NBATCH)
// ===========================================================================
__global__ void k_knn(const float* __restrict__ Q, const float* __restrict__ Cc,
                      int* __restrict__ idx, int Lq, int Lc)
{
  int t = blockIdx.x * blockDim.x + threadIdx.x;
  if (t >= Lq * NBATCH) return;
  int l = t / NBATCH, n = t % NBATCH;
  float qx = Q[(size_t)(l * NBATCH + n) * 3 + 0];
  float qy = Q[(size_t)(l * NBATCH + n) * 3 + 1];
  float qz = Q[(size_t)(l * NBATCH + n) * 3 + 2];
  float bd[KNN]; int bi[KNN];
#pragma unroll
  for (int i = 0; i < KNN; ++i) { bd[i] = 3.4e38f; bi[i] = 0; }
  for (int m = 0; m < Lc; ++m) {
    float dx = Cc[(size_t)(m * NBATCH + n) * 3 + 0] - qx;
    float dy = Cc[(size_t)(m * NBATCH + n) * 3 + 1] - qy;
    float dz = Cc[(size_t)(m * NBATCH + n) * 3 + 2] - qz;
    float d = dx * dx + dy * dy + dz * dz;
    float wmax = bd[0]; int ws = 0;
#pragma unroll
    for (int i = 1; i < KNN; ++i) if (bd[i] > wmax) { wmax = bd[i]; ws = i; }
    if (d < wmax) {
#pragma unroll
      for (int i = 0; i < KNN; ++i) if (i == ws) { bd[i] = d; bi[i] = m; }
    }
  }
  for (int k = 0; k < KNN; ++k) idx[((size_t)k * Lq + l) * NBATCH + n] = bi[k];
}

// x[r, :] = feats[idx[r], n, :]  with r = (k*L + l)*NBATCH + n
__global__ void k_gather_feats(const float* __restrict__ feats, const int* __restrict__ idx,
                               float* __restrict__ x, int L, int D)
{
  long long t = (long long)blockIdx.x * blockDim.x + threadIdx.x;
  long long total = (long long)KNN * L * NBATCH * D;
  if (t >= total) return;
  int d = (int)(t % D);
  long long r = t / D;
  int n = (int)(r % NBATCH);
  int id = idx[r];
  x[(size_t)r * D + d] = feats[((size_t)id * NBATCH + n) * D + d];
}

// pos_in[r,:] = coords[idx[r], n, :] - coords[l, n, :]
__global__ void k_pos_in(const float* __restrict__ coords, const int* __restrict__ idx,
                         float* __restrict__ pin, int L)
{
  long long r = (long long)blockIdx.x * blockDim.x + threadIdx.x;
  long long total = (long long)KNN * L * NBATCH;
  if (r >= total) return;
  int n = (int)(r % NBATCH);
  int l = (int)((r / NBATCH) % L);
  int id = idx[r];
#pragma unroll
  for (int c = 0; c < 3; ++c)
    pin[(size_t)r * 3 + c] = coords[((size_t)id * NBATCH + n) * 3 + c]
                           - coords[((size_t)l  * NBATCH + n) * 3 + c];
}

// 16-token softmax attention, one thread per (b, h, s). qkv row = (s*B+b)*3E.
__global__ void k_attn(const float* __restrict__ qkv, float* __restrict__ o, int B, int E)
{
  int t = blockIdx.x * blockDim.x + threadIdx.x;
  int total = B * HEADS * 16;
  if (t >= total) return;
  int s = t & 15; int bh = t >> 4; int h = bh % HEADS; int b = bh / HEADS;
  int dh = E / HEADS;
  float scale = rsqrtf((float)dh);
  size_t r3E = 3 * (size_t)E;
  const float* qp = qkv + ((size_t)s * B + b) * r3E + h * dh;
  float sc[16]; float mx = -3.4e38f;
#pragma unroll
  for (int tt = 0; tt < 16; ++tt) {
    const float* kp = qkv + ((size_t)tt * B + b) * r3E + E + h * dh;
    float acc = 0.f;
    for (int d = 0; d < dh; ++d) acc += qp[d] * kp[d];
    acc *= scale; sc[tt] = acc; mx = fmaxf(mx, acc);
  }
  float sum = 0.f;
#pragma unroll
  for (int tt = 0; tt < 16; ++tt) { float e = __expf(sc[tt] - mx); sc[tt] = e; sum += e; }
  float inv = 1.f / sum;
  float* op = o + ((size_t)s * B + b) * E + h * dh;
  for (int d = 0; d < dh; ++d) {
    float acc = 0.f;
#pragma unroll
    for (int tt = 0; tt < 16; ++tt)
      acc += sc[tt] * qkv[((size_t)tt * B + b) * r3E + 2 * E + h * dh + d];
    op[d] = acc * inv;
  }
}

// y = LayerNorm(a + b) * g + beta ; one wave32 per row.
__global__ __launch_bounds__(256) void k_ln_res(
    const float* __restrict__ A, const float* __restrict__ Bv,
    const float* __restrict__ g, const float* __restrict__ be,
    float* __restrict__ Y, int R, int E)
{
  int wave = threadIdx.x >> 5, lane = threadIdx.x & 31;
  int row = blockIdx.x * 8 + wave;
  if (row >= R) return;
  const float* a = A + (size_t)row * E;
  const float* b = Bv + (size_t)row * E;
  float s = 0.f, s2 = 0.f;
  for (int i = lane; i < E; i += 32) { float v = a[i] + b[i]; s += v; s2 += v * v; }
#pragma unroll
  for (int off = 16; off; off >>= 1) { s += __shfl_xor(s, off, 32); s2 += __shfl_xor(s2, off, 32); }
  float mean = s / E;
  float var = s2 / E - mean * mean;
  float inv = rsqrtf(var + 1e-5f);
  float* y = Y + (size_t)row * E;
  for (int i = lane; i < E; i += 32) {
    float v = a[i] + b[i];
    y[i] = (v - mean) * inv * g[i] + be[i];
  }
}

// out[b, d] = mean over s of x[(s*B + b)*E + d]
__global__ void k_mean_k(const float* __restrict__ x, float* __restrict__ out, int B, int E)
{
  long long t = (long long)blockIdx.x * blockDim.x + threadIdx.x;
  if (t >= (long long)B * E) return;
  int d = (int)(t % E); int b = (int)(t / E);
  float acc = 0.f;
#pragma unroll
  for (int s = 0; s < 16; ++s) acc += x[((size_t)s * B + b) * E + d];
  out[(size_t)b * E + d] = acc * (1.f / 16.f);
}

// keep_c[m,n,:] = c[m*4, n, :]
__global__ void k_keep_gather(const float* __restrict__ c, float* __restrict__ kc, int M)
{
  int t = blockIdx.x * blockDim.x + threadIdx.x;
  if (t >= M * NBATCH * 3) return;
  int cc = t % 3; int n = (t / 3) % NBATCH; int m = t / (3 * NBATCH);
  kc[t] = c[(((size_t)m * 4) * NBATCH + n) * 3 + cc];
}

// pool[m,n,d] = max_k f[idx[k,m,n], n, d]
__global__ void k_gather_max(const float* __restrict__ f, const int* __restrict__ idx,
                             float* __restrict__ pool, int M, int D)
{
  long long t = (long long)blockIdx.x * blockDim.x + threadIdx.x;
  if (t >= (long long)M * NBATCH * D) return;
  int d = (int)(t % D); int n = (int)((t / D) % NBATCH); int m = (int)(t / ((long long)D * NBATCH));
  float acc = -3.4e38f;
#pragma unroll
  for (int k = 0; k < KNN; ++k) {
    int id = idx[((size_t)k * M + m) * NBATCH + n];
    acc = fmaxf(acc, f[((size_t)id * NBATCH + n) * D + d]);
  }
  pool[t] = acc;
}

__global__ void k_zero(float* __restrict__ p, long long n)
{
  long long t = (long long)blockIdx.x * blockDim.x + threadIdx.x;
  if (t < n) p[t] = 0.f;
}

// scatter-mean accumulation for transition_up
__global__ void k_scatter_add(const float* __restrict__ pc, const int* __restrict__ idx,
                              float* __restrict__ s, float* __restrict__ cnt,
                              int Mc, int D)
{
  long long t = (long long)blockIdx.x * blockDim.x + threadIdx.x;
  long long total = (long long)KNN * Mc * NBATCH * D;
  if (t >= total) return;
  int d = (int)(t % D);
  long long r = t / D;              // (k*Mc + m)*NBATCH + n
  int n = (int)(r % NBATCH);
  int m = (int)((r / NBATCH) % Mc);
  int id = idx[r];
  atomicAdd(&s[((size_t)id * NBATCH + n) * D + d], pc[((size_t)m * NBATCH + n) * D + d]);
  if (d == 0) atomicAdd(&cnt[(size_t)id * NBATCH + n], 1.f);
}

__global__ void k_up_combine(const float* __restrict__ pf, const float* __restrict__ s,
                             const float* __restrict__ cnt, float* __restrict__ fine,
                             int L, int D)
{
  long long t = (long long)blockIdx.x * blockDim.x + threadIdx.x;
  if (t >= (long long)L * NBATCH * D) return;
  long long ln = t / D;
  float c = fmaxf(cnt[ln], 1.f);
  fine[t] = pf[t] + s[t] / c;
}

__global__ void k_add(const float* __restrict__ a, const float* __restrict__ b,
                      float* __restrict__ y, long long n)
{
  long long t = (long long)blockIdx.x * blockDim.x + threadIdx.x;
  if (t < n) y[t] = a[t] + b[t];
}

// ===========================================================================
// Host orchestration
// ===========================================================================
struct EncLayerP {
  const float *inw, *inb, *outw, *outb, *ff1w, *ff1b, *ff2w, *ff2b;
  const float *ln1g, *ln1b, *ln2g, *ln2b;
  const float *p1w, *p1b, *p2w, *p2b;   // pos-enc mlp
  int E, F;
};

extern "C" void kernel_launch(void* const* d_in, const int* in_sizes, int n_in,
                              void* d_out, int out_size, void* d_ws, size_t ws_size,
                              hipStream_t stream)
{
  (void)in_sizes; (void)n_in; (void)out_size; (void)ws_size;
  auto P = [&](int i) { return (const float*)d_in[i]; };

  const float* coords   = P(0);   // (2048, 2, 3)
  const float* features = P(1);   // (2048, 2, 16)

  // --- d_ws bump allocators -------------------------------------------------
  char* ws = (char*)d_ws;
  size_t poff = 0;
  auto pal_f = [&](size_t n) { poff = (poff + 255) & ~(size_t)255; float* p = (float*)(ws + poff); poff += n * 4; return p; };
  auto pal_i = [&](size_t n) { poff = (poff + 255) & ~(size_t)255; int*   p = (int*)(ws + poff);   poff += n * 4; return p; };

  const int L0 = 2048;
  const int Ls[4] = {2048, 512, 128, 32};
  const int Ds[4] = {64, 128, 256, 512};

  // persistent buffers
  float* feats0      = pal_f((size_t)L0 * NBATCH * 64);
  float* featsA      = pal_f((size_t)L0 * NBATCH * 64);
  float* first_feats = pal_f((size_t)L0 * NBATCH * 64);
  int*   first_idx   = pal_i((size_t)KNN * L0 * NBATCH);
  float* keepc[3]; int* nbr_idx[3]; int* attn_idx[3]; float* pool[3]; float* encout[3];
  for (int i = 0; i < 3; ++i) {
    int M = Ls[i + 1], D2 = 2 * Ds[i];
    keepc[i]    = pal_f((size_t)M * NBATCH * 3);
    nbr_idx[i]  = pal_i((size_t)KNN * M * NBATCH);
    attn_idx[i] = pal_i((size_t)KNN * M * NBATCH);
    pool[i]     = pal_f((size_t)M * NBATCH * D2);
    encout[i]   = pal_f((size_t)M * NBATCH * D2);
  }
  float* bottom_out = pal_f((size_t)32 * NBATCH * 512);
  float* fine_out[3]; float* decout[3];
  const int decL[3] = {128, 512, 2048};
  const int decD[3] = {256, 128, 64};
  for (int j = 0; j < 3; ++j) {
    fine_out[j] = pal_f((size_t)decL[j] * NBATCH * decD[j]);
    decout[j]   = pal_f((size_t)decL[j] * NBATCH * decD[j]);
  }
  const size_t arena0 = (poff + 255) & ~(size_t)255;
  size_t aoff = arena0;
  auto areset = [&]() { aoff = arena0; };
  auto aal_f  = [&](size_t n) { aoff = (aoff + 255) & ~(size_t)255; float* p = (float*)(ws + aoff); aoff += n * 4; return p; };

  // --- tiny launch helpers --------------------------------------------------
  auto gemm = [&](const float* A, const float* W, const float* b, float* C,
                  int M, int N, int K, int act, int accum) {
    dim3 g(M / 64, N / 64);
    k_gemm<<<g, 256, 0, stream>>>(A, W, b, C, M, N, K, act, accum);
  };
  auto g1d = [](long long n) { return dim3((unsigned)((n + 255) / 256)); };

  // --- param leaf indices (jax tree-flatten, dict keys sorted) --------------
  // 2..5 bottom(l1.b,l1.w,l2.b,l2.w); 6+20j dec[j]; 66+18i enc[i];
  // 120..123 first_mlp; 124..127 tail; 128..139 top_layer; 140..143 top_pos.
  auto dec_layer = [&](int j) {
    int b = 6 + 20 * j;
    EncLayerP L;
    L.ff1b = P(b + 0);  L.ff1w = P(b + 1);  L.ff2b = P(b + 2);  L.ff2w = P(b + 3);
    L.inb  = P(b + 4);  L.inw  = P(b + 5);
    L.ln1b = P(b + 6);  L.ln1g = P(b + 7);  L.ln2b = P(b + 8);  L.ln2g = P(b + 9);
    L.outb = P(b + 10); L.outw = P(b + 11);
    L.p1b  = P(b + 12); L.p1w  = P(b + 13); L.p2b = P(b + 14); L.p2w = P(b + 15);
    L.E = decD[j]; L.F = 2 * decD[j];
    return L;
  };
  auto enc_layer = [&](int i) {
    int b = 66 + 18 * i;
    EncLayerP L;
    L.ff1b = P(b + 2);  L.ff1w = P(b + 3);  L.ff2b = P(b + 4);  L.ff2w = P(b + 5);
    L.inb  = P(b + 6);  L.inw  = P(b + 7);
    L.ln1b = P(b + 8);  L.ln1g = P(b + 9);  L.ln2b = P(b + 10); L.ln2g = P(b + 11);
    L.outb = P(b + 12); L.outw = P(b + 13);
    L.p1b  = P(b + 14); L.p1w  = P(b + 15); L.p2b = P(b + 16); L.p2w = P(b + 17);
    L.E = 2 * Ds[i]; L.F = 4 * Ds[i];
    return L;
  };
  EncLayerP topL;
  topL.ff1b = P(128); topL.ff1w = P(129); topL.ff2b = P(130); topL.ff2w = P(131);
  topL.inb  = P(132); topL.inw  = P(133);
  topL.ln1b = P(134); topL.ln1g = P(135); topL.ln2b = P(136); topL.ln2g = P(137);
  topL.outb = P(138); topL.outw = P(139);
  topL.p1b  = P(140); topL.p1w  = P(141); topL.p2b = P(142); topL.p2w = P(143);
  topL.E = 64; topL.F = 2048;

  // --- attention block ------------------------------------------------------
  auto attn_block = [&](const EncLayerP& Lp, const float* cl, const float* fl,
                        const int* idx, int L, bool fused_top, float* out) {
    const int E = Lp.E, F = Lp.F;
    const int R = KNN * L * NBATCH;      // rows (multiple of 64)
    const int B = L * NBATCH;
    areset();
    float* pin  = aal_f((size_t)R * 3);
    float* posh = aal_f((size_t)R * E);
    float* x    = aal_f((size_t)R * E);
    float* qkv  = aal_f((size_t)R * 3 * E);
    float* ao   = aal_f((size_t)R * E);
    float* xo   = aal_f((size_t)R * E);
    float* x1   = aal_f((size_t)R * E);
    float* ffy  = aal_f((size_t)R * E);
    float* x2   = aal_f((size_t)R * E);
    float* ffh  = fused_top ? nullptr : aal_f((size_t)R * F);

    k_pos_in<<<g1d((long long)R), 256, 0, stream>>>(cl, idx, pin, L);
    gemm(pin, Lp.p1w, Lp.p1b, posh, R, E, 3, /*relu*/1, 0);
    k_gather_feats<<<g1d((long long)R * E), 256, 0, stream>>>(fl, idx, x, L, E);
    gemm(posh, Lp.p2w, Lp.p2b, x, R, E, E, 0, /*accum*/1);     // x += pos_enc
    gemm(x, Lp.inw, Lp.inb, qkv, R, 3 * E, E, 0, 0);
    k_attn<<<g1d((long long)B * HEADS * 16), 256, 0, stream>>>(qkv, ao, B, E);
    gemm(ao, Lp.outw, Lp.outb, xo, R, E, E, 0, 0);
    k_ln_res<<<dim3(R / 8), 256, 0, stream>>>(x, xo, Lp.ln1g, Lp.ln1b, x1, R, E);
    if (fused_top) {
      k_ffn_top<<<dim3(R / 64), 256, 0, stream>>>(x1, Lp.ff1w, Lp.ff1b, Lp.ff2w, Lp.ff2b, ffy, R);
    } else {
      gemm(x1, Lp.ff1w, Lp.ff1b, ffh, R, F, E, /*relu*/1, 0);
      gemm(ffh, Lp.ff2w, Lp.ff2b, ffy, R, E, F, 0, 0);
    }
    k_ln_res<<<dim3(R / 8), 256, 0, stream>>>(x1, ffy, Lp.ln2g, Lp.ln2b, x2, R, E);
    k_mean_k<<<g1d((long long)B * E), 256, 0, stream>>>(x2, out, B, E);
  };

  // ========================= forward pass ==================================
  // tail mlp: 16 -> 64 relu -> 64
  {
    areset();
    float* h = aal_f((size_t)L0 * NBATCH * 64);
    gemm(features, P(125), P(124), h, L0 * NBATCH, 64, 16, 1, 0);
    gemm(h, P(127), P(126), feats0, L0 * NBATCH, 64, 64, 0, 0);
  }
  // first_idx + top attention block (E=64, F=2048 fused)
  k_knn<<<g1d((long long)L0 * NBATCH), 256, 0, stream>>>(coords, coords, first_idx, L0, L0);
  attn_block(topL, coords, feats0, first_idx, L0, /*fused*/true, featsA);
  // first_feats = mlp(first_mlp, featsA)
  {
    areset();
    float* h = aal_f((size_t)L0 * NBATCH * 64);
    gemm(featsA, P(121), P(120), h, L0 * NBATCH, 64, 64, 1, 0);
    gemm(h, P(123), P(122), first_feats, L0 * NBATCH, 64, 64, 0, 0);
  }

  // encoder blocks
  const float* c_lvl = coords;
  const float* f_lvl = featsA;
  const float* skip_c[3]; const float* skip_f[3];
  for (int i = 0; i < 3; ++i) {
    const int L = Ls[i], M = Ls[i + 1], db = Ds[i], D2 = 2 * db;
    const int encb = 66 + 18 * i;
    skip_c[i] = c_lvl; skip_f[i] = f_lvl;
    // transition_down
    k_keep_gather<<<g1d((long long)M * NBATCH * 3), 256, 0, stream>>>(c_lvl, keepc[i], M);
    k_knn<<<g1d((long long)M * NBATCH), 256, 0, stream>>>(keepc[i], c_lvl, nbr_idx[i], M, L);
    areset();
    float* f = aal_f((size_t)L * NBATCH * D2);
    gemm(f_lvl, P(encb + 1), P(encb + 0), f, L * NBATCH, D2, db, /*silu*/2, 0);
    k_gather_max<<<g1d((long long)M * NBATCH * D2), 256, 0, stream>>>(f, nbr_idx[i], pool[i], M, D2);
    // self attention at coarse level
    k_knn<<<g1d((long long)M * NBATCH), 256, 0, stream>>>(keepc[i], keepc[i], attn_idx[i], M, M);
    attn_block(enc_layer(i), keepc[i], pool[i], attn_idx[i], M, false, encout[i]);
    c_lvl = keepc[i]; f_lvl = encout[i];
  }

  // bottom mlp: 512 -> 512 relu -> 512  (rows = 64)
  {
    areset();
    float* h = aal_f((size_t)32 * NBATCH * 512);
    gemm(f_lvl, P(3), P(2), h, 32 * NBATCH, 512, 512, 1, 0);
    gemm(h, P(5), P(4), bottom_out, 32 * NBATCH, 512, 512, 0, 0);
  }

  // decoder blocks  (dec j works at skip level 2-j)
  const float* coarse = bottom_out;
  for (int j = 0; j < 3; ++j) {
    const int sk = 2 - j;                 // enc level index of the skip
    const int Lf = Ls[sk];                // fine point count
    const int Mc = Ls[sk + 1];            // coarse point count
    const int db = Ds[sk];                // fine feature dim (== decD[j])
    const int b = 6 + 20 * j;
    const float* sc = skip_c[sk];
    const float* sf = skip_f[sk];
    const int* nbr = nbr_idx[sk];
    const int* aidx = (j < 2) ? attn_idx[sk - 1] : first_idx;
    // transition_up
    areset();
    float* pc  = aal_f((size_t)Mc * NBATCH * db);
    float* pf  = aal_f((size_t)Lf * NBATCH * db);
    float* s   = aal_f((size_t)Lf * NBATCH * db);
    float* cnt = aal_f((size_t)Lf * NBATCH);
    gemm(coarse, P(b + 17), P(b + 16), pc, Mc * NBATCH, db, 2 * db, /*silu*/2, 0);
    gemm(sf,     P(b + 19), P(b + 18), pf, Lf * NBATCH, db, db,     /*silu*/2, 0);
    k_zero<<<g1d((long long)Lf * NBATCH * db), 256, 0, stream>>>(s, (long long)Lf * NBATCH * db);
    k_zero<<<g1d((long long)Lf * NBATCH), 256, 0, stream>>>(cnt, (long long)Lf * NBATCH);
    k_scatter_add<<<g1d((long long)KNN * Mc * NBATCH * db), 256, 0, stream>>>(pc, nbr, s, cnt, Mc, db);
    k_up_combine<<<g1d((long long)Lf * NBATCH * db), 256, 0, stream>>>(pf, s, cnt, fine_out[j], Lf, db);
    // attention block at fine level
    attn_block(dec_layer(j), sc, fine_out[j], aidx, Lf, false, decout[j]);
    coarse = decout[j];
  }

  // output = feats + first_feats
  k_add<<<g1d((long long)L0 * NBATCH * 64), 256, 0, stream>>>(
      decout[2], first_feats, (float*)d_out, (long long)L0 * NBATCH * 64);
}
